// MHA_49400713838761
// MI455X (gfx1250) — matrix operand order repaired and checked by reading, not compiled, not run on hardware
//
#include <hip/hip_runtime.h>

// ---------------------------------------------------------------------------
// MI455X (gfx1250) multi-head attention, bf16 WMMA path, f32 accumulate.
// v_wmma_f32_16x16x32_bf16 + Tensor Data Mover (tensor_load_to_lds) staging
// (async-to-LDS fallback), register-ping-pong pipelined GEMMs.
// ---------------------------------------------------------------------------

typedef __bf16 bf16;
typedef __attribute__((ext_vector_type(8)))  __bf16 v8bf;
typedef __attribute__((ext_vector_type(16))) __bf16 v16bf;
typedef __attribute__((ext_vector_type(8)))  float  v8f;

#define HEADS  12
#define DH     64
#define DMODEL 768
#define NSEQ   1024
#define BATCH  8

#if defined(__has_builtin)
#if __has_builtin(__builtin_amdgcn_tensor_load_to_lds) && \
    __has_builtin(__builtin_amdgcn_s_wait_tensorcnt)
#define USE_TDM 1
#endif
#endif
#ifndef USE_TDM
#define USE_TDM 0
#endif

__device__ __forceinline__ v8bf ld8(const bf16* p) { return *(const v8bf*)p; }

__device__ __forceinline__ v16bf join16(v8bf lo, v8bf hi) {
  v16bf r;
#pragma unroll
  for (int i = 0; i < 8; ++i) { r[i] = lo[i]; r[i + 8] = hi[i]; }
  return r;
}

__device__ __forceinline__ v8f wmma_bf16(v16bf a, v16bf b, v8f c) {
  return __builtin_amdgcn_wmma_f32_16x16x32_bf16(
      false, a, false, b, (short)0, c, false, false);
}

// ---- async fallback helpers (ASYNCcnt) -----------------------------------
__device__ __forceinline__ void async_cp16(const bf16* lds_dst,
                                           const bf16* gsrc) {
  uint32_t lds_off = (uint32_t)(uintptr_t)lds_dst;  // addr[31:0] = LDS offset
  asm volatile("global_load_async_to_lds_b128 %0, %1, off"
               :: "v"(lds_off), "v"(gsrc) : "memory");
}
__device__ __forceinline__ void wait_async_all() {
  asm volatile("s_wait_asynccnt 0x0" ::: "memory");
}
__device__ __forceinline__ void wait_async_16() {
  asm volatile("s_wait_asynccnt 0x10" ::: "memory");
}
__device__ __forceinline__ void wait_ds() {
  asm volatile("s_wait_dscnt 0x0" ::: "memory");
}
__device__ __forceinline__ void compiler_fence() {
  asm volatile("" ::: "memory");
}

// ---- Tensor Data Mover: 2D tile (2-byte elements) global -> LDS ----------
#if USE_TDM
typedef __attribute__((ext_vector_type(4))) unsigned int u32x4;
typedef __attribute__((ext_vector_type(4))) int          i32x4;
typedef __attribute__((ext_vector_type(8))) int          i32x8;

// D# per CDNA5 ISA 8.3/8.4: group0 = {flags, lds_addr, global_addr, type},
// group1 = {data_size, tensor_dim0/1, tile_dim0/1, tensor_dim0_stride}.
__device__ __forceinline__ void tdm_load_2d(uint32_t lds_off, const void* gsrc,
                                            uint32_t tensor_d0,
                                            uint32_t tensor_d1,
                                            uint32_t stride_d0,
                                            uint32_t tile_d0,
                                            uint32_t tile_d1) {
  const unsigned long long ga = (unsigned long long)(uintptr_t)gsrc;
  u32x4 g0;
  g0[0] = 1u;                                   // count=1 (valid user D#)
  g0[1] = lds_off;                              // lds_addr (bytes)
  g0[2] = (uint32_t)ga;                         // global_addr[31:0]
  g0[3] = (uint32_t)((ga >> 32) & 0x01FFFFFFu)  // global_addr[56:32]
          | (2u << 30);                         // type = 2 ("image")
  i32x8 g1;
  g1[0] = (int)(1u << 16);                      // data_size=1 -> 2 bytes/elem
  g1[1] = (int)((tensor_d0 & 0xFFFFu) << 16);   // tensor_dim0[15:0] @bit48
  g1[2] = (int)((tensor_d0 >> 16) | ((tensor_d1 & 0xFFFFu) << 16));
  g1[3] = (int)((tensor_d1 >> 16) | ((tile_d0 & 0xFFFFu) << 16));
  g1[4] = (int)(tile_d1 & 0xFFFFu);             // tile_dim1 (tile_dim2=0)
  g1[5] = (int)stride_d0;                       // tensor_dim0_stride[31:0]
  g1[6] = 0;                                    // stride hi / dim1_stride lo
  g1[7] = 0;
  i32x4 z4 = {0, 0, 0, 0};
#if __clang_major__ >= 23
  i32x8 z8 = {0, 0, 0, 0, 0, 0, 0, 0};
  __builtin_amdgcn_tensor_load_to_lds(g0, g1, z4, z4, z8, 0);
#else
  __builtin_amdgcn_tensor_load_to_lds(g0, g1, z4, z4, 0);
#endif
}
// Builtin requires a constant-integer argument -> template parameter.
template <int N>
__device__ __forceinline__ void wait_tensor() {
  __builtin_amdgcn_s_wait_tensorcnt(N);
  compiler_fence();
}
#endif  // USE_TDM

// ---------------------------------------------------------------------------
// Prep kernels: fp32 -> bf16 convert / transpose-convert
// ---------------------------------------------------------------------------
__global__ void cvt_f32_bf16_kernel(const float* __restrict__ src,
                                    bf16* __restrict__ dst, int n) {
  int i = blockIdx.x * blockDim.x + threadIdx.x;
  if (i < n) dst[i] = (bf16)src[i];
}

__global__ void transpose_f32_bf16_kernel(const float* __restrict__ src,
                                          bf16* __restrict__ dst,
                                          int rows, int cols) {
  long i = (long)blockIdx.x * blockDim.x + threadIdx.x;
  long total = (long)rows * cols;
  if (i >= total) return;
  int c = (int)(i % cols);
  int k = (int)(i / cols);
  dst[(size_t)c * rows + k] = (bf16)src[i];
}

// ---------------------------------------------------------------------------
// GEMM core: block = 4 waves, tile 16 rows x 256 cols (wave: 16x64).
// A strip (16x768) staged in LDS by TDM (or async DMA); B fragments
// ping-pong double-buffered with a fully unrolled K loop (no reg copies).
// ---------------------------------------------------------------------------
__device__ __forceinline__ void gemm_core(const bf16* __restrict__ A,
                                          const bf16* __restrict__ Bt,
                                          bf16* Atile, int row0, int col0,
                                          int lm, int g, v8f acc[4]) {
#if USE_TDM
  if (threadIdx.x < 32) {  // one TDM per block (wave 0), EXEC-independent
    tdm_load_2d((uint32_t)(uintptr_t)Atile, A + (size_t)row0 * DMODEL,
                /*tensor_d0=*/DMODEL, /*tensor_d1=*/16,
                /*stride_d0=*/DMODEL, /*tile_d0=*/DMODEL, /*tile_d1=*/16);
    wait_tensor<0>();
  }
#else
  const int tid = threadIdx.x;
#pragma unroll
  for (int i = 0; i < 12; ++i) {
    int c = tid + i * 128;
    int row = c / 96, seg = c % 96;  // 96 16B-chunks per 1536B row
    async_cp16(&Atile[row * DMODEL + seg * 8],
               A + (size_t)(row0 + row) * DMODEL + seg * 8);
  }
  wait_async_all();
#endif
  __syncthreads();

#pragma unroll
  for (int t = 0; t < 4; ++t)
#pragma unroll
    for (int r = 0; r < 8; ++r) acc[t][r] = 0.0f;

  const bf16* Bp[4];
#pragma unroll
  for (int t = 0; t < 4; ++t)
    Bp[t] = Bt + (size_t)(col0 + t * 16 + lm) * DMODEL + 16 * g;

  const bf16* Arow = &Atile[lm * DMODEL];

  // Ping-pong buffers with constant indices; full unroll kills all copies.
  v8bf b0[2][4], b1[2][4];
#pragma unroll
  for (int t = 0; t < 4; ++t) {
    b0[0][t] = ld8(Bp[t]);
    b1[0][t] = ld8(Bp[t] + 8);
  }

#pragma unroll
  for (int kc = 0; kc < DMODEL; kc += 64) {
    // Prefetch chunk kc+32 into buffer 1 while computing chunk kc (buffer 0).
#pragma unroll
    for (int t = 0; t < 4; ++t) {
      b0[1][t] = ld8(Bp[t] + kc + 32);
      b1[1][t] = ld8(Bp[t] + kc + 40);
    }
    {
      v16bf a = join16(ld8(Arow + kc + 16 * g), ld8(Arow + kc + 8 + 16 * g));
#pragma unroll
      for (int t = 0; t < 4; ++t)
        acc[t] = wmma_bf16(a, join16(b0[0][t], b1[0][t]), acc[t]);
    }
    // Prefetch chunk kc+64 into buffer 0 while computing kc+32 (buffer 1).
    if (kc + 64 < DMODEL) {
#pragma unroll
      for (int t = 0; t < 4; ++t) {
        b0[0][t] = ld8(Bp[t] + kc + 64);
        b1[0][t] = ld8(Bp[t] + kc + 72);
      }
    }
    {
      v16bf a = join16(ld8(Arow + kc + 32 + 16 * g),
                       ld8(Arow + kc + 40 + 16 * g));
#pragma unroll
      for (int t = 0; t < 4; ++t)
        acc[t] = wmma_bf16(a, join16(b0[1][t], b1[1][t]), acc[t]);
    }
  }
}

// ---------------------------------------------------------------------------
// QKV projection -> scatter into Q[B,h,N,dh], K[B,h,N,dh], Vt[B,h,dh,N]
// ---------------------------------------------------------------------------
__global__ void __launch_bounds__(128)
gemm_qkv_kernel(const bf16* __restrict__ A, const bf16* __restrict__ Bt,
                bf16* __restrict__ Q, bf16* __restrict__ K,
                bf16* __restrict__ Vt) {
  __shared__ __align__(16) bf16 Atile[16 * DMODEL];  // 24 KB
  const int lane = threadIdx.x & 31;
  const int wave = threadIdx.x >> 5;
  const int lm   = lane & 15;
  const int g    = lane >> 4;
  const int row0 = blockIdx.x * 16;
  const int col0 = blockIdx.y * 256 + wave * 64;

  v8f acc[4];
  gemm_core(A, Bt, Atile, row0, col0, lm, g, acc);

#pragma unroll
  for (int t = 0; t < 4; ++t) {
    const int col   = col0 + t * 16 + lm;
    const int which = col / DMODEL;
    const int d     = col % DMODEL;
    const int h     = d / DH;
    const int hd    = d % DH;
#pragma unroll
    for (int r = 0; r < 8; ++r) {
      const int row = row0 + r + 8 * g;
      const int bb  = row / NSEQ;
      const int nn  = row % NSEQ;
      const size_t head = (size_t)(bb * HEADS + h);
      const bf16 val = (bf16)acc[t][r];
      if (which == 0)
        Q[head * NSEQ * DH + (size_t)nn * DH + hd] = val;
      else if (which == 1)
        K[head * NSEQ * DH + (size_t)nn * DH + hd] = val;
      else
        Vt[head * DH * NSEQ + (size_t)hd * NSEQ + nn] = val;
    }
  }
}

// ---------------------------------------------------------------------------
// Output projection: out = attnb (8192x768) * w_out + b_out (f32)
// ---------------------------------------------------------------------------
__global__ void __launch_bounds__(128)
gemm_out_kernel(const bf16* __restrict__ A, const bf16* __restrict__ Bt,
                const float* __restrict__ bias, float* __restrict__ out) {
  __shared__ __align__(16) bf16 Atile[16 * DMODEL];
  const int lane = threadIdx.x & 31;
  const int wave = threadIdx.x >> 5;
  const int lm   = lane & 15;
  const int g    = lane >> 4;
  const int row0 = blockIdx.x * 16;
  const int col0 = blockIdx.y * 256 + wave * 64;

  v8f acc[4];
  gemm_core(A, Bt, Atile, row0, col0, lm, g, acc);

#pragma unroll
  for (int t = 0; t < 4; ++t) {
    const int col  = col0 + t * 16 + lm;
    const float bv = bias[col];
#pragma unroll
    for (int r = 0; r < 8; ++r) {
      const int row = row0 + r + 8 * g;
      out[(size_t)row * DMODEL + col] = acc[t][r] + bv;
    }
  }
}

// ---------------------------------------------------------------------------
// Flash attention: one wave per (b, h, 16-row Q tile). 32-key chunks.
// K/V chunks double-buffered in LDS via TDM (one 2D tile per matrix per
// chunk, TENSORcnt-synchronized) or async-to-LDS fallback.
// ---------------------------------------------------------------------------
__global__ void __launch_bounds__(32)
attn_kernel(const bf16* __restrict__ Q, const bf16* __restrict__ K,
            const bf16* __restrict__ Vt, bf16* __restrict__ attnb) {
  const int lane = threadIdx.x & 31;
  const int lm   = lane & 15;
  const int g    = lane >> 4;
  const int qt   = blockIdx.x;
  const int h    = blockIdx.y;
  const int b    = blockIdx.z;

  const size_t head = (size_t)(b * HEADS + h) * NSEQ * DH;
  const bf16* Qbase = Q + head + (size_t)(qt * 16 + lm) * DH;
  const bf16* Khead = K + head;   // [N][dh]
  const bf16* Vhead = Vt + head;  // [dh][N]

  __shared__ __align__(16) bf16 Kbuf[2][32 * DH];  // 2 x 4 KB
  __shared__ __align__(16) bf16 Vbuf[2][DH * 32];  // 2 x 4 KB
  __shared__ __align__(16) bf16 Pl[16 * 32];       // 1 KB P tile

  auto stage = [&](int buf, int kc) {
#if USE_TDM
    // K tile: 32 key-rows x 64 dh (row-contiguous, stride 64 elems)
    tdm_load_2d((uint32_t)(uintptr_t)&Kbuf[buf][0], Khead + (size_t)kc * DH,
                /*tensor_d0=*/DH, /*tensor_d1=*/32,
                /*stride_d0=*/DH, /*tile_d0=*/DH, /*tile_d1=*/32);
    // V tile: 64 dh-rows x 32 keys (stride NSEQ elems)
    tdm_load_2d((uint32_t)(uintptr_t)&Vbuf[buf][0], Vhead + kc,
                /*tensor_d0=*/NSEQ, /*tensor_d1=*/DH,
                /*stride_d0=*/NSEQ, /*tile_d0=*/32, /*tile_d1=*/DH);
#else
#pragma unroll
    for (int i = 0; i < 8; ++i) {
      int c = lane + i * 32;
      int row = c >> 3, seg = c & 7;
      async_cp16(&Kbuf[buf][row * DH + seg * 8],
                 Khead + (size_t)(kc + row) * DH + seg * 8);
    }
#pragma unroll
    for (int i = 0; i < 8; ++i) {
      int c = lane + i * 32;
      int row = c >> 2, seg = c & 3;
      async_cp16(&Vbuf[buf][row * 32 + seg * 8],
                 Vhead + (size_t)row * NSEQ + kc + seg * 8);
    }
#endif
  };

  v16bf aq0 = join16(ld8(Qbase + 16 * g), ld8(Qbase + 8 + 16 * g));
  v16bf aq1 = join16(ld8(Qbase + 32 + 16 * g), ld8(Qbase + 40 + 16 * g));

  float mrow[8], lrow[8];
  v8f o[4];
#pragma unroll
  for (int r = 0; r < 8; ++r) { mrow[r] = -3.0e30f; lrow[r] = 0.0f; }
#pragma unroll
  for (int t = 0; t < 4; ++t)
#pragma unroll
    for (int r = 0; r < 8; ++r) o[t][r] = 0.0f;

  const float scale = 0.036084391824351615f;  // 768^-0.5

  stage(0, 0);

  for (int kc = 0; kc < NSEQ; kc += 32) {
    const int cur = (kc >> 5) & 1;
    if (kc + 32 < NSEQ) {
      stage(cur ^ 1, kc + 32);  // overlap next chunk's DMA with compute
#if USE_TDM
      wait_tensor<2>();         // in-order: current chunk's 2 tiles done
#else
      wait_async_16();
#endif
    } else {
#if USE_TDM
      wait_tensor<0>();
#else
      wait_async_all();
#endif
    }

    // ---- S = Q K^T over 32 keys (two 16-col tiles), K from LDS ----
    v8f s[2];
#pragma unroll
    for (int nt = 0; nt < 2; ++nt) {
      const bf16* Krow = &Kbuf[cur][(nt * 16 + lm) * DH];
      v16bf bk0 = join16(ld8(Krow + 16 * g), ld8(Krow + 16 * g + 8));
      v16bf bk1 = join16(ld8(Krow + 32 + 16 * g), ld8(Krow + 40 + 16 * g));
      v8f c;
#pragma unroll
      for (int r = 0; r < 8; ++r) c[r] = 0.0f;
      c = wmma_bf16(aq0, bk0, c);
      c = wmma_bf16(aq1, bk1, c);
      s[nt] = c;
    }

    // ---- online softmax ----
#pragma unroll
    for (int r = 0; r < 8; ++r) {
      float v0 = s[0][r] * scale;
      float v1 = s[1][r] * scale;
      float cm = fmaxf(v0, v1);
      cm = fmaxf(cm, __shfl_xor(cm, 1, 32));
      cm = fmaxf(cm, __shfl_xor(cm, 2, 32));
      cm = fmaxf(cm, __shfl_xor(cm, 4, 32));
      cm = fmaxf(cm, __shfl_xor(cm, 8, 32));
      float mnew  = fmaxf(mrow[r], cm);
      float alpha = __expf(mrow[r] - mnew);
      float p0 = __expf(v0 - mnew);
      float p1 = __expf(v1 - mnew);
      mrow[r] = mnew;
      float ps = p0 + p1;
      ps += __shfl_xor(ps, 1, 32);
      ps += __shfl_xor(ps, 2, 32);
      ps += __shfl_xor(ps, 4, 32);
      ps += __shfl_xor(ps, 8, 32);
      lrow[r] = lrow[r] * alpha + ps;
#pragma unroll
      for (int t = 0; t < 4; ++t) o[t][r] *= alpha;
      const int m_ = r + 8 * g;
      Pl[m_ * 32 + lm]      = (bf16)p0;
      Pl[m_ * 32 + 16 + lm] = (bf16)p1;
    }
    wait_ds();  // cross-lane P visibility before A-layout reload

    // ---- O += P * V ----
    const bf16* Prow = &Pl[lm * 32];
    v16bf pa = join16(ld8(Prow + 16 * g), ld8(Prow + 8 + 16 * g));
#pragma unroll
    for (int t = 0; t < 4; ++t) {
      const bf16* Vrow = &Vbuf[cur][(t * 16 + lm) * 32 + 16 * g];
      v16bf vb = join16(ld8(Vrow), ld8(Vrow + 8));
      o[t] = wmma_bf16(pa, vb, o[t]);
    }
    wait_ds();  // drain P reads before next iteration overwrites Pl
  }

#pragma unroll
  for (int r = 0; r < 8; ++r) lrow[r] = 1.0f / lrow[r];
  bf16* outb = attnb + (size_t)b * NSEQ * DMODEL;
#pragma unroll
  for (int t = 0; t < 4; ++t) {
#pragma unroll
    for (int r = 0; r < 8; ++r) {
      const int n   = qt * 16 + r + 8 * g;
      const int col = h * DH + t * 16 + lm;
      outb[(size_t)n * DMODEL + col] = (bf16)(o[t][r] * lrow[r]);
    }
  }
}

// ---------------------------------------------------------------------------
// Host launcher
// ---------------------------------------------------------------------------
extern "C" void kernel_launch(void* const* d_in, const int* in_sizes, int n_in,
                              void* d_out, int out_size, void* d_ws,
                              size_t ws_size, hipStream_t stream) {
  (void)in_sizes; (void)n_in; (void)out_size; (void)ws_size;
  const float* x     = (const float*)d_in[0];
  const float* w_qkv = (const float*)d_in[1];
  const float* w_out = (const float*)d_in[2];
  const float* b_out = (const float*)d_in[3];
  float* out = (float*)d_out;

  char* ws = (char*)d_ws;
  size_t o = 0;
  auto take = [&](size_t bytes) -> char* {
    char* p = ws + o;
    o += (bytes + 255) & ~(size_t)255;
    return p;
  };
  const size_t rowsA = (size_t)BATCH * NSEQ;
  bf16* xb    = (bf16*)take(rowsA * DMODEL * sizeof(bf16));
  bf16* wqkvT = (bf16*)take((size_t)3 * DMODEL * DMODEL * sizeof(bf16));
  bf16* woutT = (bf16*)take((size_t)DMODEL * DMODEL * sizeof(bf16));
  bf16* Q     = (bf16*)take(rowsA * DMODEL * sizeof(bf16));
  bf16* K     = (bf16*)take(rowsA * DMODEL * sizeof(bf16));
  bf16* Vt    = (bf16*)take(rowsA * DMODEL * sizeof(bf16));
  bf16* attnb = (bf16*)take(rowsA * DMODEL * sizeof(bf16));

  {
    int n = (int)(rowsA * DMODEL);
    cvt_f32_bf16_kernel<<<(n + 255) / 256, 256, 0, stream>>>(x, xb, n);
  }
  {
    long total = (long)DMODEL * (3 * DMODEL);
    transpose_f32_bf16_kernel<<<(unsigned)((total + 255) / 256), 256, 0,
                                stream>>>(w_qkv, wqkvT, DMODEL, 3 * DMODEL);
  }
  {
    long total = (long)DMODEL * DMODEL;
    transpose_f32_bf16_kernel<<<(unsigned)((total + 255) / 256), 256, 0,
                                stream>>>(w_out, woutT, DMODEL, DMODEL);
  }

  gemm_qkv_kernel<<<dim3((unsigned)(rowsA / 16), (3 * DMODEL) / 256), 128, 0,
                    stream>>>(xb, wqkvT, Q, K, Vt);
  attn_kernel<<<dim3(NSEQ / 16, HEADS, BATCH), 32, 0, stream>>>(Q, K, Vt,
                                                                attnb);
  gemm_out_kernel<<<dim3((unsigned)(rowsA / 16), DMODEL / 256), 128, 0,
                    stream>>>(attnb, woutT, b_out, out);
}
